// Flow_34660386078870
// MI455X (gfx1250) — matrix operand order repaired and checked by reading, not compiled, not run on hardware
//
#include <hip/hip_runtime.h>
#include <hip/hip_bf16.h>

typedef __attribute__((ext_vector_type(16))) _Float16 v16h;
typedef __attribute__((ext_vector_type(8)))  _Float16 v8h;
typedef __attribute__((ext_vector_type(2)))  __fp16   f16x2;
typedef __attribute__((ext_vector_type(8)))  float    v8f;

#define B_   128
#define N_   512
#define IN_  128
#define L_   64
#define H_   64
#define C_   10
#define FL_  4

__device__ __forceinline__ float wave_sum32(float v) {
  #pragma unroll
  for (int o = 16; o > 0; o >>= 1) v += __shfl_xor(v, o, 32);
  return v;
}

__device__ __forceinline__ unsigned int pk2(float a, float b) {
  f16x2 p = __builtin_amdgcn_cvt_pkrtz(a, b);
  return __builtin_bit_cast(unsigned int, p);
}

// ---------------------------------------------------------------------------
// One-shot: WsT16[f,c] = f16(Ws[c,f])  (64x128, feature-major, c contiguous)
// ---------------------------------------------------------------------------
__global__ void k_prep(const float* __restrict__ Ws, _Float16* __restrict__ WsT16) {
  int f = blockIdx.x;        // 0..63
  int c = threadIdx.x;       // 0..127
  WsT16[f * IN_ + c] = (_Float16)Ws[c * L_ + f];
}

// ---------------------------------------------------------------------------
// Fused over the mandatory full read of gs:
//   rdeg[b,j] = 1 / sum_k gs[b,j,k]     (f32 row sums)
//   gs16      = f16(gs)                 (halves later passes' gs traffic)
// ---------------------------------------------------------------------------
__global__ void k_rdeg(const float* __restrict__ gs, float* __restrict__ rdeg,
                       _Float16* __restrict__ gs16) {
  int wave = threadIdx.x >> 5, lane = threadIdx.x & 31;
  int row = blockIdx.x * 8 + wave;            // row in [0, B*N)
  const float* p = gs + (size_t)row * N_;
  _Float16*    q = gs16 + (size_t)row * N_;
  float s = 0.f;
  #pragma unroll
  for (int t = 0; t < 4; ++t) {
    int o = lane * 4 + 128 * t;               // contiguous float4 per lane
    float4 v = *(const float4*)&p[o];
    s += (v.x + v.y) + (v.z + v.w);
    uint2 u; u.x = pk2(v.x, v.y); u.y = pk2(v.z, v.w);
    *(uint2*)&q[o] = u;
  }
  s = wave_sum32(s);
  if (lane == 0) rdeg[row] = 1.0f / s;
}

// ---------------------------------------------------------------------------
// X16T[b,f,n] = f16( rdeg[b,n] * sum_c hs[b,n,c] * Ws[c,f] )
// WMMA f16 16x16x32; 128 threads (4 waves); 64x64 output tile per block.
// ---------------------------------------------------------------------------
__global__ __launch_bounds__(128) void k_x(const float* __restrict__ hs,
                                           const _Float16* __restrict__ WsT16,
                                           const float* __restrict__ rdeg,
                                           _Float16* __restrict__ X16T) {
  __shared__ alignas(64) _Float16 As[64 * 32];
  __shared__ alignas(64) _Float16 Bs[64 * 32];
  int b  = blockIdx.x >> 3;
  int i0 = (blockIdx.x & 7) * 64;
  int t = threadIdx.x, w = t >> 5, lane = t & 31;
  v8f acc[4];
  #pragma unroll
  for (int tc = 0; tc < 4; ++tc)
    #pragma unroll
    for (int e = 0; e < 8; ++e) acc[tc][e] = 0.f;

  const int mfrag = lane & 15;
  const int base0 = (lane < 16) ? 0 : 16;
  const int kbase = (lane < 16) ? 0 : 16;

  int bF[2], bK[2];
  #pragma unroll
  for (int r = 0; r < 2; ++r) {
    int c = t + 128 * r;                 // 8-half chunk of 64x32 B tile
    bF[r] = c >> 2; bK[r] = (c & 3) * 8;
  }

  for (int kk = 0; kk < IN_ / 32; ++kk) {
    float4 aReg[4];
    #pragma unroll
    for (int r = 0; r < 4; ++r) {
      int s = t + 128 * r;               // float4 segment
      int m = s >> 3, k = (s & 7) * 4;
      aReg[r] = *(const float4*)&hs[(size_t)(b * N_ + i0 + m) * IN_ + kk * 32 + k];
    }
    v8h bReg[2];
    #pragma unroll
    for (int r = 0; r < 2; ++r)
      bReg[r] = *(const v8h*)&WsT16[bF[r] * IN_ + kk * 32 + bK[r]];

    #pragma unroll
    for (int r = 0; r < 4; ++r) {
      int s = t + 128 * r;
      int m = s >> 3, k = (s & 7) * 4;
      uint2 u;
      u.x = pk2(aReg[r].x, aReg[r].y);
      u.y = pk2(aReg[r].z, aReg[r].w);
      *(uint2*)&As[m * 32 + k] = u;
    }
    #pragma unroll
    for (int r = 0; r < 2; ++r) *(v8h*)&Bs[bF[r] * 32 + bK[r]] = bReg[r];
    __syncthreads();

    int m = w * 16 + mfrag;
    v8h alo = *(const v8h*)&As[m * 32 + base0];
    v8h ahi = *(const v8h*)&As[m * 32 + base0 + 8];
    v16h a;
    #pragma unroll
    for (int i = 0; i < 8; ++i) { a[i] = alo[i]; a[i + 8] = ahi[i]; }
    #pragma unroll
    for (int tc = 0; tc < 4; ++tc) {
      v16h bf = *(const v16h*)&Bs[(tc * 16 + mfrag) * 32 + kbase];
      acc[tc] = __builtin_amdgcn_wmma_f32_16x16x32_f16(false, a, false, bf,
                                                       (short)0, acc[tc], false, false);
    }
    __syncthreads();
  }

  int colL  = lane & 15;
  int rbase = i0 + w * 16 + ((lane >= 16) ? 8 : 0);
  float4 rd0 = *(const float4*)&rdeg[b * N_ + rbase];
  float4 rd1 = *(const float4*)&rdeg[b * N_ + rbase + 4];
  #pragma unroll
  for (int tc = 0; tc < 4; ++tc) {
    uint4 u;
    u.x = pk2(acc[tc][0] * rd0.x, acc[tc][1] * rd0.y);
    u.y = pk2(acc[tc][2] * rd0.z, acc[tc][3] * rd0.w);
    u.z = pk2(acc[tc][4] * rd1.x, acc[tc][5] * rd1.y);
    u.w = pk2(acc[tc][6] * rd1.z, acc[tc][7] * rd1.w);
    *(uint4*)&X16T[(size_t)(b * 64 + tc * 16 + colL) * N_ + rbase] = u;
  }
}

// ---------------------------------------------------------------------------
// h[b,i,f] = relu( sum_k gs16[b,i,k] * X16T[b,f,k] + bs[f] )
// Tiles are filled with GLOBAL_LOAD_ASYNC_TO_LDS_B128 (ASYNCcnt path): the
// copy bypasses VGPRs, next tile's fills overlap the current WMMA block, and
// the wave pays only s_wait_asynccnt after the barrier.
// ---------------------------------------------------------------------------
__global__ __launch_bounds__(128) void k_h(const _Float16* __restrict__ gs16,
                                           const _Float16* __restrict__ X16T,
                                           const float* __restrict__ bs,
                                           float* __restrict__ h) {
  __shared__ alignas(64) _Float16 As[2][64 * 32];
  __shared__ alignas(64) _Float16 Bs[2][64 * 32];
  int b  = blockIdx.x >> 3;
  int i0 = (blockIdx.x & 7) * 64;
  int t = threadIdx.x, w = t >> 5, lane = t & 31;
  v8f acc[4];
  #pragma unroll
  for (int tc = 0; tc < 4; ++tc)
    #pragma unroll
    for (int e = 0; e < 8; ++e) acc[tc][e] = 0.f;

  const int mfrag = lane & 15;
  const int base0 = (lane < 16) ? 0 : 16;
  const int kbase = (lane < 16) ? 0 : 16;

  // staging geometry: both tiles are 64x32 halves = 256 chunks of 8 halves
  const _Float16* aG[2];
  const _Float16* bG[2];
  unsigned aL[2][2], bL[2][2];             // [buffer][chunk] LDS byte address
  #pragma unroll
  for (int r = 0; r < 2; ++r) {
    int c = t + 128 * r;
    int m = c >> 2, k = (c & 3) * 8;
    aG[r] = gs16 + (size_t)(b * N_ + i0 + m) * N_ + k;
    bG[r] = X16T + (size_t)(b * 64 + m) * N_ + k;
    #pragma unroll
    for (int q = 0; q < 2; ++q) {
      aL[q][r] = (unsigned)(uintptr_t)&As[q][m * 32 + k];
      bL[q][r] = (unsigned)(uintptr_t)&Bs[q][m * 32 + k];
    }
  }

  const int S = N_ / 32;                   // 16 K-steps

  // prologue: async-fill buffer 0 with tile 0
  #pragma unroll
  for (int r = 0; r < 2; ++r)
    asm volatile("global_load_async_to_lds_b128 %0, %1, off"
                 :: "v"(aL[0][r]), "v"(aG[r]) : "memory");
  #pragma unroll
  for (int r = 0; r < 2; ++r)
    asm volatile("global_load_async_to_lds_b128 %0, %1, off"
                 :: "v"(bL[0][r]), "v"(bG[r]) : "memory");
  asm volatile("s_wait_asynccnt 0x0" ::: "memory");
  __syncthreads();

  for (int kk = 0; kk < S; ++kk) {
    int p = kk & 1;

    // issue async fills of buffer 1-p with tile kk+1 (overlaps WMMAs below;
    // safe: trailing barrier of iter kk-1 retired all reads of buffer 1-p)
    if (kk + 1 < S) {
      #pragma unroll
      for (int r = 0; r < 2; ++r)
        asm volatile("global_load_async_to_lds_b128 %0, %1, off"
                     :: "v"(aL[1 - p][r]), "v"(aG[r] + (kk + 1) * 32) : "memory");
      #pragma unroll
      for (int r = 0; r < 2; ++r)
        asm volatile("global_load_async_to_lds_b128 %0, %1, off"
                     :: "v"(bL[1 - p][r]), "v"(bG[r] + (kk + 1) * 32) : "memory");
    }

    // fragments + WMMA from buffer p
    int m = w * 16 + mfrag;
    v8h alo = *(const v8h*)&As[p][m * 32 + base0];
    v8h ahi = *(const v8h*)&As[p][m * 32 + base0 + 8];
    v16h a;
    #pragma unroll
    for (int i = 0; i < 8; ++i) { a[i] = alo[i]; a[i + 8] = ahi[i]; }
    #pragma unroll
    for (int tc = 0; tc < 4; ++tc) {
      v16h bf = *(const v16h*)&Bs[p][(tc * 16 + mfrag) * 32 + kbase];
      acc[tc] = __builtin_amdgcn_wmma_f32_16x16x32_f16(false, a, false, bf,
                                                       (short)0, acc[tc], false, false);
    }
    __syncthreads();                       // all waves done reading buffer p

    if (kk + 1 < S) {
      asm volatile("s_wait_asynccnt 0x0" ::: "memory");   // own fills landed
      __syncthreads();                     // everyone's fills landed
    }
  }

  int rbase = i0 + w * 16 + ((lane >= 16) ? 8 : 0);
  int colL  = lane & 15;
  #pragma unroll
  for (int tc = 0; tc < 4; ++tc)
    #pragma unroll
    for (int v = 0; v < 8; ++v) {
      int col = tc * 16 + colL;
      float val = acc[tc][v] + bs[col];
      h[(size_t)(b * N_ + rbase + v) * L_ + col] = fmaxf(val, 0.f);
    }
}

// ---------------------------------------------------------------------------
// smu[b,n] = rdeg[b,n] * (h[b,n,:] . Wmu),  slv likewise. Wave per row.
// ---------------------------------------------------------------------------
__global__ void k_mulv(const float* __restrict__ h, const float* __restrict__ Wmu,
                       const float* __restrict__ Wlv, const float* __restrict__ rdeg,
                       float* __restrict__ smu, float* __restrict__ slv) {
  int wave = threadIdx.x >> 5, lane = threadIdx.x & 31;
  int row = blockIdx.x * 8 + wave;
  const float* hp = h + (size_t)row * L_;
  float h0 = hp[lane], h1 = hp[lane + 32];
  float dmu = h0 * Wmu[lane] + h1 * Wmu[lane + 32];
  float dlv = h0 * Wlv[lane] + h1 * Wlv[lane + 32];
  dmu = wave_sum32(dmu); dlv = wave_sum32(dlv);
  if (lane == 0) { float r = rdeg[row]; smu[row] = r * dmu; slv[row] = r * dlv; }
}

// ---------------------------------------------------------------------------
// Fused: mu/logvar matvecs over gs16 + reparam + planar flow + KL + gate.
// ---------------------------------------------------------------------------
__global__ void k_gv(const _Float16* __restrict__ gs16, const float* __restrict__ smu,
                     const float* __restrict__ slv, const float* __restrict__ eps,
                     const float* __restrict__ bmu, const float* __restrict__ blv,
                     const float* __restrict__ beta, const float* __restrict__ fw,
                     const float* __restrict__ fb, const float* __restrict__ fs,
                     const float* __restrict__ rdeg, float* __restrict__ zs,
                     float* __restrict__ szs, float* __restrict__ klp) {
  __shared__ float red[8];
  int wave = threadIdx.x >> 5, lane = threadIdx.x & 31;
  int row = blockIdx.x * 8 + wave;           // row = b*N + i
  int b = row >> 9;
  const _Float16* gp = gs16 + (size_t)row * N_;
  const float* smb = smu + b * N_;
  const float* slb = slv + b * N_;
  float tmu = 0.f, tlv = 0.f;
  #pragma unroll
  for (int t = 0; t < 2; ++t) {
    int o = lane * 8 + 256 * t;              // contiguous v8h per lane
    v8h g8 = *(const v8h*)&gp[o];
    #pragma unroll
    for (int e = 0; e < 8; ++e) {
      float gv = (float)g8[e];
      tmu += gv * smb[o + e];
      tlv += gv * slb[o + e];
    }
  }
  tmu = wave_sum32(tmu); tlv = wave_sum32(tlv);
  if (lane == 0) {
    float mu = fmaxf(tmu + bmu[0], 0.f);
    float lv = fmaxf(tlv + blv[0], 0.f);
    float sigma = expf(0.5f * lv);
    float z0 = eps[row] * sigma + mu;
    float z = z0, lj = 0.f;
    #pragma unroll
    for (int k = 0; k < FL_; ++k) {
      float tt  = tanhf(fw[k] * z + fb[k]);
      float det = 1.f + fs[k] * fw[k] * (1.f - tt * tt);
      lj += logf(fabsf(det) + 1e-7f);
      z  += fs[k] * tt;
    }
    float u = (z0 - mu) / sigma;
    float logq = -0.5f * u * u - logf(2.5f * sigma);
    float logp = -0.5f * z * z - logf(2.5f);
    float zv = 1.f / (1.f + expf(-beta[0] * z));
    zs[row]  = zv;
    szs[row] = rdeg[row] * zv;
    red[wave] = logq - lj - logp;
  }
  __syncthreads();
  if (threadIdx.x == 0) {
    float s = 0.f;
    #pragma unroll
    for (int i = 0; i < 8; ++i) s += red[i];
    klp[blockIdx.x] = s;
  }
}

// ---------------------------------------------------------------------------
// dz[b,i] = sum_j gs16[b,i,j]*szs[b,j]; d = relu(dz*Wdec + bdec); mse partials.
// ---------------------------------------------------------------------------
__global__ void k_dec(const _Float16* __restrict__ gs16, const float* __restrict__ szs,
                      const float* __restrict__ h, const float* __restrict__ Wdec,
                      const float* __restrict__ bdec, float* __restrict__ msep) {
  __shared__ float red[8];
  int wave = threadIdx.x >> 5, lane = threadIdx.x & 31;
  int row = blockIdx.x * 8 + wave;
  int b = row >> 9;
  const _Float16* gp = gs16 + (size_t)row * N_;
  const float* sz = szs + b * N_;
  float dz = 0.f;
  #pragma unroll
  for (int t = 0; t < 2; ++t) {
    int o = lane * 8 + 256 * t;
    v8h g8 = *(const v8h*)&gp[o];
    #pragma unroll
    for (int e = 0; e < 8; ++e) dz += (float)g8[e] * sz[o + e];
  }
  dz = wave_sum32(dz);                        // butterfly: full sum in every lane
  const float* hp = h + (size_t)row * L_;
  float m = 0.f;
  #pragma unroll
  for (int u = 0; u < 2; ++u) {
    int l = lane + 32 * u;
    float d = fmaxf(dz * Wdec[l] + bdec[l], 0.f);
    float diff = hp[l] - d;
    m += diff * diff;
  }
  m = wave_sum32(m);
  if (lane == 0) red[wave] = m;
  __syncthreads();
  if (threadIdx.x == 0) {
    float s = 0.f;
    #pragma unroll
    for (int i = 0; i < 8; ++i) s += red[i];
    msep[blockIdx.x] = s;
  }
}

// hg[b,l] = sum_n h[b,n,l] * zs[b,n] ; one 64-thread block per batch
__global__ void k_hg(const float* __restrict__ h, const float* __restrict__ zs,
                     float* __restrict__ hg) {
  int b = blockIdx.x, l = threadIdx.x;
  const float* hp = h + (size_t)b * N_ * L_;
  const float* zp = zs + b * N_;
  float s = 0.f;
  for (int n = 0; n < N_; ++n) s += hp[n * L_ + l] * zp[n];
  hg[b * L_ + l] = s;
}

// classifier head + log_softmax + per-batch nll/acc partials
__global__ void k_cls(const float* __restrict__ hg, const float* __restrict__ W1,
                      const float* __restrict__ b1, const float* __restrict__ W2,
                      const float* __restrict__ b2, const int* __restrict__ labels,
                      float* __restrict__ nllp, float* __restrict__ accp) {
  __shared__ float xs[H_];
  __shared__ float lg[C_];
  int b = blockIdx.x, t = threadIdx.x;
  const float* hgb = hg + b * L_;
  float s = 0.f;
  for (int f = 0; f < L_; ++f) s += hgb[f] * W1[f * H_ + t];
  xs[t] = fmaxf(s + b1[t], 0.f);
  __syncthreads();
  if (t < C_) {
    float s2 = 0.f;
    for (int l = 0; l < H_; ++l) s2 += xs[l] * W2[l * C_ + t];
    lg[t] = s2 + b2[t];
  }
  __syncthreads();
  if (t == 0) {
    float mx = lg[0]; int pred = 0;
    for (int c = 1; c < C_; ++c) if (lg[c] > mx) { mx = lg[c]; pred = c; }
    float lse = 0.f;
    for (int c = 0; c < C_; ++c) lse += expf(lg[c] - mx);
    lse = logf(lse);
    int lab = labels[b];
    nllp[b] = -(lg[lab] - mx - lse);
    accp[b] = (pred == lab) ? 1.f : 0.f;
  }
}

// deterministic final reduction: fixed strided accumulation + LDS tree
__global__ void k_final(const float* __restrict__ klp, const float* __restrict__ msep,
                        const float* __restrict__ nllp, const float* __restrict__ accp,
                        float* __restrict__ out) {
  __shared__ float red[256];
  int t = threadIdx.x;
  float s_kl = 0.f, s_mse = 0.f, s_nll = 0.f, s_acc = 0.f;
  for (int i = t; i < 8192; i += 256) { s_kl += klp[i]; s_mse += msep[i]; }
  for (int i = t; i < B_;   i += 256) { s_nll += nllp[i]; s_acc += accp[i]; }

  red[t] = s_kl; __syncthreads();
  for (int o = 128; o > 0; o >>= 1) { if (t < o) red[t] += red[t + o]; __syncthreads(); }
  float kl_sum = red[0]; __syncthreads();

  red[t] = s_mse; __syncthreads();
  for (int o = 128; o > 0; o >>= 1) { if (t < o) red[t] += red[t + o]; __syncthreads(); }
  float mse_sum = red[0]; __syncthreads();

  red[t] = s_nll; __syncthreads();
  for (int o = 128; o > 0; o >>= 1) { if (t < o) red[t] += red[t + o]; __syncthreads(); }
  float nll_sum = red[0]; __syncthreads();

  red[t] = s_acc; __syncthreads();
  for (int o = 128; o > 0; o >>= 1) { if (t < o) red[t] += red[t + o]; __syncthreads(); }
  float acc_sum = red[0];

  if (t == 0) {
    float loss = nll_sum / (float)B_
               + mse_sum / (float)((size_t)B_ * N_ * L_)
               + kl_sum  / (float)(B_ * N_);
    out[0] = loss;
    out[1] = acc_sum / (float)B_;
  }
}

extern "C" void kernel_launch(void* const* d_in, const int* in_sizes, int n_in,
                              void* d_out, int out_size, void* d_ws, size_t ws_size,
                              hipStream_t stream) {
  (void)in_sizes; (void)n_in; (void)out_size; (void)ws_size;
  const float* gs   = (const float*)d_in[0];
  const float* hs   = (const float*)d_in[1];
  const int*   labs = (const int*)  d_in[2];
  const float* eps  = (const float*)d_in[3];
  const float* Ws   = (const float*)d_in[4];
  const float* bs   = (const float*)d_in[5];
  const float* Wmu  = (const float*)d_in[6];
  const float* bmu  = (const float*)d_in[7];
  const float* Wlv  = (const float*)d_in[8];
  const float* blv  = (const float*)d_in[9];
  const float* Wdec = (const float*)d_in[10];
  const float* bdec = (const float*)d_in[11];
  const float* W1   = (const float*)d_in[12];
  const float* b1   = (const float*)d_in[13];
  const float* W2   = (const float*)d_in[14];
  const float* b2   = (const float*)d_in[15];
  const float* beta = (const float*)d_in[16];
  const float* fw   = (const float*)d_in[17];
  const float* fb   = (const float*)d_in[18];
  const float* fs   = (const float*)d_in[19];
  float* out = (float*)d_out;

  char* p = (char*)d_ws;
  auto alloc = [&](size_t bytes) -> char* {
    char* r = p; p += (bytes + 255) & ~(size_t)255; return r;
  };
  _Float16* gs16  = (_Float16*)alloc(sizeof(_Float16) * (size_t)B_ * N_ * N_);
  float*    rdeg  = (float*)   alloc(sizeof(float)    * (size_t)B_ * N_);
  _Float16* X16T  = (_Float16*)alloc(sizeof(_Float16) * (size_t)B_ * N_ * L_);
  float*    h     = (float*)   alloc(sizeof(float)    * (size_t)B_ * N_ * L_);
  _Float16* WsT16 = (_Float16*)alloc(sizeof(_Float16) * (size_t)L_ * IN_);
  float*    smu   = (float*)   alloc(sizeof(float)    * (size_t)B_ * N_);
  float*    slv   = (float*)   alloc(sizeof(float)    * (size_t)B_ * N_);
  float*    zs    = (float*)   alloc(sizeof(float)    * (size_t)B_ * N_);
  float*    szs   = (float*)   alloc(sizeof(float)    * (size_t)B_ * N_);
  float*    hg    = (float*)   alloc(sizeof(float)    * (size_t)B_ * L_);
  float*    klp   = (float*)   alloc(sizeof(float)    * 8192);
  float*    msep  = (float*)   alloc(sizeof(float)    * 8192);
  float*    nllp  = (float*)   alloc(sizeof(float)    * B_);
  float*    accp  = (float*)   alloc(sizeof(float)    * B_);

  k_prep <<<L_, IN_, 0, stream>>>(Ws, WsT16);
  k_rdeg <<<(B_ * N_) / 8, 256, 0, stream>>>(gs, rdeg, gs16);
  k_x    <<<B_ * (N_ / 64), 128, 0, stream>>>(hs, WsT16, rdeg, X16T);
  k_h    <<<B_ * (N_ / 64), 128, 0, stream>>>(gs16, X16T, bs, h);
  k_mulv <<<(B_ * N_) / 8, 256, 0, stream>>>(h, Wmu, Wlv, rdeg, smu, slv);
  k_gv   <<<(B_ * N_) / 8, 256, 0, stream>>>(gs16, smu, slv, eps, bmu, blv, beta,
                                             fw, fb, fs, rdeg, zs, szs, klp);
  k_dec  <<<(B_ * N_) / 8, 256, 0, stream>>>(gs16, szs, h, Wdec, bdec, msep);
  k_hg   <<<B_, 64, 0, stream>>>(h, zs, hg);
  k_cls  <<<B_, 64, 0, stream>>>(hg, W1, b1, W2, b2, labs, nllp, accp);
  k_final<<<1, 256, 0, stream>>>(klp, msep, nllp, accp, out);
}
